// KoLeoLoss_64450279244468
// MI455X (gfx1250) — compile-verified, matrix operand order checked by reading, not run on hardware
//
#include <hip/hip_runtime.h>
#include <hip/hip_bf16.h>

// ---------------------------------------------------------------------------
// KoLeo loss for x[8192,1024] fp32 on gfx1250 (MI455X).
// Compute-bound on the 8192x8192x1024 similarity GEMM (137 GFLOP); inputs fit
// in L2 (32MB << 192MB).  GEMM runs on v_wmma_f32_16x16x32_bf16 with a
// 3xBF16 (hi/lo) split for ~fp32 accuracy.  Each wave computes a 32x64
// register tile (2 row tiles x 4 col tiles): every B load now feeds 6 WMMAs,
// giving ~33 FLOP/B from L2 (vs ~19 before) so the matrix pipes, not the L2
// read path, are the limiter.  S is never materialized: tiles reduce to
// per-row packed (value,col) keys merged with atomicMax (deterministic;
// tie-break = lowest column, matching argmax).
// ---------------------------------------------------------------------------

typedef __bf16 v4bf  __attribute__((ext_vector_type(4)));
typedef __bf16 v8bf  __attribute__((ext_vector_type(8)));
typedef __bf16 v16bf __attribute__((ext_vector_type(16)));
typedef float  v8f   __attribute__((ext_vector_type(8)));

#define KL_N   8192
#define KL_D   1024
#define KL_EPS 1e-8f

// ---------------------------------------------------------------------------
// Kernel 1: row L2-normalize, split into bf16 hi/lo, store inv-norm,
//           reset argmax keys.  One block (256 threads) per row.
// ---------------------------------------------------------------------------
__global__ void __launch_bounds__(256)
kl_normalize_split(const float* __restrict__ x,
                   __bf16* __restrict__ hi, __bf16* __restrict__ lo,
                   float* __restrict__ invn,
                   unsigned long long* __restrict__ keys) {
  __shared__ float red[256];
  const int r = blockIdx.x;
  const int t = threadIdx.x;
  const float4* xr4 = (const float4*)(x + (size_t)r * KL_D);

  float4 a = xr4[t];                       // 4 elements per thread (256*4=1024)
  float ss = a.x * a.x + a.y * a.y + a.z * a.z + a.w * a.w;
  red[t] = ss;
  __syncthreads();
  for (int s = 128; s > 0; s >>= 1) {
    if (t < s) red[t] += red[t + s];
    __syncthreads();
  }
  if (t == 0) {
    float inv = 1.0f / fmaxf(sqrtf(red[0]), KL_EPS);
    red[0] = inv;
    invn[r] = inv;
    keys[r] = 0ull;                        // reset argmax key every call
  }
  __syncthreads();
  const float inv = red[0];

  float xn[4] = {a.x * inv, a.y * inv, a.z * inv, a.w * inv};
  v4bf h, l;
#pragma unroll
  for (int c = 0; c < 4; ++c) {
    __bf16 hb = (__bf16)xn[c];
    h[c] = hb;
    l[c] = (__bf16)(xn[c] - (float)hb);
  }
  const size_t base = (size_t)r * KL_D + t * 4;   // 8-byte aligned
  *(v4bf*)(hi + base) = h;
  *(v4bf*)(lo + base) = l;
}

// ---------------------------------------------------------------------------
// WMMA A/B tile loader.  16-bit A-matrix 16x32 layout (wave32):
//   lanes 0-15 : M = lane,    K = {0..7} and {16..23}
//   lanes 16-31: M = lane-16, K = {8..15} and {24..31}
// i.e. per lane: two contiguous 16B chunks at K = 8*half and K = 16+8*half.
// For S = X*X^T the B (KxN) tile column n is row n of X, with the mirrored
// per-lane layout, so the same loader serves both operands.
// ---------------------------------------------------------------------------
__device__ __forceinline__ v16bf kl_load_ab(const __bf16* __restrict__ p,
                                            int row, int k, int half) {
  const __bf16* q = p + (size_t)row * KL_D + k + half * 8;
  v8bf c0 = *(const v8bf*)q;          // global_load_b128
  v8bf c1 = *(const v8bf*)(q + 16);   // global_load_b128
  v16bf r;
#pragma unroll
  for (int i = 0; i < 8; ++i) { r[i] = c0[i]; r[i + 8] = c1[i]; }
  return r;
}

__device__ __forceinline__ unsigned kl_ord(float f) {
  unsigned u = __float_as_uint(f);
  return (u & 0x80000000u) ? ~u : (u | 0x80000000u);   // order-preserving map
}

// ---------------------------------------------------------------------------
// Kernel 2: tiled X*X^T with per-row running argmax.
// 8 waves/block; each wave owns a 32x64 output tile (row strip ti of 32 rows,
// col strip tjs of 64 cols) and loops K=0..1024 in steps of 32.
// Per K-step: 4 A loads-pairs (2 row tiles x hi/lo) reused across 4 column
// tiles; each B (hi,lo) pair feeds 6 WMMAs (2 row tiles x 3 for the bf16
// hi/lo product expansion  (hi+lo)(hi+lo) ~= hi*hi + hi*lo + lo*hi).
// ---------------------------------------------------------------------------
__global__ void __launch_bounds__(256)
kl_gemm_argmax(const __bf16* __restrict__ hi, const __bf16* __restrict__ lo,
               unsigned long long* __restrict__ keys) {
  const int lane = threadIdx.x & 31;
  const int wave = threadIdx.x >> 5;
  const int gw   = blockIdx.x * 8 + wave;      // 256 * 128 = 32768 waves
  const int ti   = gw >> 7;                    // [0,256): 32-row strip
  const int tjs  = gw & 127;                   // [0,128): 64-col strip
  const int rbase  = ti * 32;
  const int cbase0 = tjs * 64;
  const int nsub = lane & 15;
  const int half = lane >> 4;
  const int arow0 = rbase + nsub;
  const int arow1 = rbase + 16 + nsub;

  v8f acc[2][4];
#pragma unroll
  for (int rt = 0; rt < 2; ++rt)
#pragma unroll
    for (int j = 0; j < 4; ++j)
#pragma unroll
      for (int v = 0; v < 8; ++v) acc[rt][j][v] = 0.0f;

  for (int k = 0; k < KL_D; k += 32) {
    v16bf aH0 = kl_load_ab(hi, arow0, k, half);
    v16bf aL0 = kl_load_ab(lo, arow0, k, half);
    v16bf aH1 = kl_load_ab(hi, arow1, k, half);
    v16bf aL1 = kl_load_ab(lo, arow1, k, half);
#pragma unroll
    for (int j = 0; j < 4; ++j) {
      const int crow = cbase0 + j * 16 + nsub;
      v16bf bH = kl_load_ab(hi, crow, k, half);
      v16bf bL = kl_load_ab(lo, crow, k, half);
      acc[0][j] = __builtin_amdgcn_wmma_f32_16x16x32_bf16(
          false, aH0, false, bH, (short)0, acc[0][j], false, false);
      acc[0][j] = __builtin_amdgcn_wmma_f32_16x16x32_bf16(
          false, aH0, false, bL, (short)0, acc[0][j], false, false);
      acc[0][j] = __builtin_amdgcn_wmma_f32_16x16x32_bf16(
          false, aL0, false, bH, (short)0, acc[0][j], false, false);
      acc[1][j] = __builtin_amdgcn_wmma_f32_16x16x32_bf16(
          false, aH1, false, bH, (short)0, acc[1][j], false, false);
      acc[1][j] = __builtin_amdgcn_wmma_f32_16x16x32_bf16(
          false, aH1, false, bL, (short)0, acc[1][j], false, false);
      acc[1][j] = __builtin_amdgcn_wmma_f32_16x16x32_bf16(
          false, aL1, false, bH, (short)0, acc[1][j], false, false);
    }
  }

  // C/D layout: lane holds column n = lane&15; VGPR v holds row v + 8*half.
  // Per output row: combine 4 col tiles locally, butterfly-max over the 16
  // lanes of this half, then one atomicMax per row for this strip.
#pragma unroll
  for (int rt = 0; rt < 2; ++rt) {
#pragma unroll
    for (int v = 0; v < 8; ++v) {
      const int row = rbase + rt * 16 + v + 8 * half;
      unsigned long long best = 0ull;
#pragma unroll
      for (int j = 0; j < 4; ++j) {
        const int col = cbase0 + j * 16 + nsub;
        float val = acc[rt][j][v];
        if (col == row) val = -1.0f;             // mask self-similarity
        unsigned long long key =
            ((unsigned long long)kl_ord(val) << 32) | (unsigned)(~col);
        best = best > key ? best : key;          // ties -> lowest col
      }
#pragma unroll
      for (int off = 1; off < 16; off <<= 1) {
        unsigned long long o = __shfl_xor(best, off, 16);
        best = best > o ? best : o;
      }
      if (nsub == 0) atomicMax(&keys[row], best);
    }
  }
}

// ---------------------------------------------------------------------------
// Kernel 3: per-row log(||xn_r - xn_j + eps||_2 + eps).  One block per row.
// ---------------------------------------------------------------------------
__global__ void __launch_bounds__(256)
kl_dist_log(const float* __restrict__ x, const float* __restrict__ invn,
            const unsigned long long* __restrict__ keys,
            float* __restrict__ logs) {
  __shared__ float red[256];
  const int r = blockIdx.x;
  const int t = threadIdx.x;
  const int j = (int)(~(unsigned)(keys[r] & 0xffffffffull));
  const float ir = invn[r];
  const float ij = invn[j];
  const float4* xr = (const float4*)(x + (size_t)r * KL_D);
  const float4* xj = (const float4*)(x + (size_t)j * KL_D);
  float4 a = xr[t];
  float4 b = xj[t];
  float d0 = a.x * ir - b.x * ij + KL_EPS;
  float d1 = a.y * ir - b.y * ij + KL_EPS;
  float d2 = a.z * ir - b.z * ij + KL_EPS;
  float d3 = a.w * ir - b.w * ij + KL_EPS;
  red[t] = d0 * d0 + d1 * d1 + d2 * d2 + d3 * d3;
  __syncthreads();
  for (int s = 128; s > 0; s >>= 1) {
    if (t < s) red[t] += red[t + s];
    __syncthreads();
  }
  if (t == 0) logs[r] = logf(sqrtf(red[0]) + KL_EPS);
}

// ---------------------------------------------------------------------------
// Kernel 4: loss = -mean(logs).  Single block, deterministic reduction.
// ---------------------------------------------------------------------------
__global__ void __launch_bounds__(256)
kl_final_reduce(const float* __restrict__ logs, float* __restrict__ out) {
  __shared__ float red[256];
  const int t = threadIdx.x;
  float s = 0.0f;
  for (int i = t; i < KL_N; i += 256) s += logs[i];
  red[t] = s;
  __syncthreads();
  for (int st = 128; st > 0; st >>= 1) {
    if (t < st) red[t] += red[t + st];
    __syncthreads();
  }
  if (t == 0) out[0] = -red[0] / (float)KL_N;
}

// ---------------------------------------------------------------------------
extern "C" void kernel_launch(void* const* d_in, const int* in_sizes, int n_in,
                              void* d_out, int out_size, void* d_ws,
                              size_t ws_size, hipStream_t stream) {
  (void)in_sizes; (void)n_in; (void)out_size; (void)ws_size;
  const float* x = (const float*)d_in[0];

  char* ws = (char*)d_ws;
  const size_t mat_bf16 = (size_t)KL_N * KL_D * sizeof(__bf16);  // 16 MB
  __bf16* hi = (__bf16*)(ws);
  __bf16* lo = (__bf16*)(ws + mat_bf16);
  float*  invn = (float*)(ws + 2 * mat_bf16);                    // 32 KB
  unsigned long long* keys =
      (unsigned long long*)(ws + 2 * mat_bf16 + (size_t)KL_N * 4);   // 64 KB
  float* logs = (float*)(ws + 2 * mat_bf16 + (size_t)KL_N * 4 +
                         (size_t)KL_N * 8);                      // 32 KB
  float* out = (float*)d_out;

  kl_normalize_split<<<KL_N, 256, 0, stream>>>(x, hi, lo, invn, keys);
  // 256 row strips * 128 col strips = 32768 waves = 4096 blocks of 8 waves
  kl_gemm_argmax<<<4096, 256, 0, stream>>>(hi, lo, keys);
  kl_dist_log<<<KL_N, 256, 0, stream>>>(x, invn, keys, logs);
  kl_final_reduce<<<1, 256, 0, stream>>>(logs, out);
}
